// MedicalVisionTransformer_82205674045708
// MI455X (gfx1250) — compile-verified, matrix operand order checked by reading, not run on hardware
//
#include <hip/hip_runtime.h>

// ---------------------------------------------------------------------------
// MedicalVisionTransformer forward for MI455X (gfx1250, wave32, WMMA bf16)
// ---------------------------------------------------------------------------

typedef __attribute__((ext_vector_type(16))) __bf16 v16bf;
typedef __attribute__((ext_vector_type(8)))  float  v8f;

constexpr int cL  = 12;
constexpr int cH  = 768;
constexpr int cFF = 3072;
constexpr int cNH = 12;
constexpr int cDH = 64;
constexpr int cB  = 64;
constexpr int cR  = 29;
constexpr int cT  = 30;          // R + 1
constexpr int cD  = 14;
constexpr int cE  = 15;
constexpr int cLR = 8;
constexpr int cK  = 384;
constexpr int cM  = cB * cT;     // 1920 token rows
constexpr int cBET = cB * cE * cT; // 28800 expert rows
constexpr float cSC  = 2.0f;     // 16/8
constexpr float cEPS = 1e-5f;

__device__ __forceinline__ float gelu_exact(float v) {
  return 0.5f * v * (1.0f + erff(v * 0.70710678118654752f));
}

// ---------------------------------------------------------------------------
// fp32 -> bf16 conversion (weights are converted once per launch)
// ---------------------------------------------------------------------------
__global__ void k_cvt(const float* __restrict__ in, __bf16* __restrict__ out, long n) {
  long i = (long)blockIdx.x * blockDim.x + threadIdx.x;
  long stride = (long)gridDim.x * blockDim.x;
  for (; i < n; i += stride) out[i] = (__bf16)in[i];
}

// ---------------------------------------------------------------------------
// x = concat(cls, region_features)
// ---------------------------------------------------------------------------
__global__ void k_concat(const float* __restrict__ reg, const float* __restrict__ cls,
                         float* __restrict__ x) {
  int row = blockIdx.x;           // over B*T
  int b = row / cT, t = row % cT;
  for (int h = threadIdx.x; h < cH; h += blockDim.x)
    x[(size_t)row * cH + h] = (t == 0) ? cls[h]
                                       : reg[((size_t)b * cR + (t - 1)) * cH + h];
}

// ---------------------------------------------------------------------------
// generic LayerNorm over `width` (g/b optional), fp32 and/or bf16 outputs
// ---------------------------------------------------------------------------
__global__ void k_ln(const float* __restrict__ in, const float* __restrict__ g,
                     const float* __restrict__ b, float* __restrict__ outf,
                     __bf16* __restrict__ outb, int width) {
  __shared__ float red[2][8];
  int row = blockIdx.x;
  const float* x = in + (size_t)row * width;
  float s = 0.f, ss = 0.f;
  for (int i = threadIdx.x; i < width; i += blockDim.x) {
    float v = x[i]; s += v; ss += v * v;
  }
  for (int o = 16; o > 0; o >>= 1) { s += __shfl_xor(s, o, 32); ss += __shfl_xor(ss, o, 32); }
  int wv = threadIdx.x >> 5, ln = threadIdx.x & 31;
  if (ln == 0) { red[0][wv] = s; red[1][wv] = ss; }
  __syncthreads();
  if (threadIdx.x == 0) {
    float a = 0.f, c = 0.f;
    for (int q = 0; q < 8; ++q) { a += red[0][q]; c += red[1][q]; }
    red[0][0] = a; red[1][0] = c;
  }
  __syncthreads();
  float mean = red[0][0] / width;
  float var  = red[1][0] / width - mean * mean;
  float rstd = rsqrtf(var + cEPS);
  for (int i = threadIdx.x; i < width; i += blockDim.x) {
    float v = (x[i] - mean) * rstd;
    if (g) v = v * g[i] + (b ? b[i] : 0.f);
    if (outf) outf[(size_t)row * width + i] = v;
    if (outb) outb[(size_t)row * width + i] = (__bf16)v;
  }
}

// ---------------------------------------------------------------------------
// WMMA bf16 GEMM:  C[M,N] = A[M,K] * W[N,K]^T (+bias)(+gelu)(+Cadd)
// Fragment layout (v_wmma_f32_16x16x32_bf16):
//   lane l: row = l%16, kbase = (l>>4)*8; VGPR0-3 = K[kbase..kbase+7],
//   VGPR4-7 = K[kbase+16..kbase+23]  -> two contiguous b128 loads.
// Each wave computes one 16(M) x 64(N) strip (A fragment reused 4x).
// ---------------------------------------------------------------------------
__device__ __forceinline__ v16bf load_frag(const __bf16* __restrict__ base, int ld,
                                           int row0, int k0, int lane) {
  const __bf16* p = base + (size_t)(row0 + (lane & 15)) * ld + k0 + ((lane >> 4) << 3);
  union { uint4 u[2]; v16bf v; } f;
  f.u[0] = *reinterpret_cast<const uint4*>(p);
  f.u[1] = *reinterpret_cast<const uint4*>(p + 16);
  return f.v;
}

__global__ void k_gemm(const __bf16* __restrict__ A, int lda,
                       const __bf16* __restrict__ W, int ldb,
                       const float* __restrict__ bias,
                       const float* __restrict__ Cadd,
                       float* __restrict__ C, __bf16* __restrict__ Cbf, int ldc,
                       int Mt, int Nt4, int Ksize, int fuse) {
  int wave = threadIdx.x >> 5;
  int lane = threadIdx.x & 31;
  int wid  = blockIdx.x * (blockDim.x >> 5) + wave;
  if (wid >= Mt * Nt4) return;
  int tn4 = wid % Nt4;
  int tm  = wid / Nt4;

  v8f acc[4];
  #pragma unroll
  for (int s = 0; s < 4; ++s)
    #pragma unroll
    for (int r = 0; r < 8; ++r) acc[s][r] = 0.f;

  for (int k0 = 0; k0 < Ksize; k0 += 32) {
    v16bf af = load_frag(A, lda, tm * 16, k0, lane);
    if (k0 + 32 < Ksize) {
      const __bf16* pf = A + (size_t)(tm * 16 + (lane & 15)) * lda + (k0 + 32);
      __builtin_prefetch(pf, 0, 0);        // global_prefetch_b8
    }
    #pragma unroll
    for (int s = 0; s < 4; ++s) {
      v16bf bfrag = load_frag(W, ldb, (tn4 * 4 + s) * 16, k0, lane);
      acc[s] = __builtin_amdgcn_wmma_f32_16x16x32_bf16(
          false, af, false, bfrag, (short)0, acc[s], false, false);
    }
  }

  int mbase = tm * 16 + ((lane >> 4) << 3);
  #pragma unroll
  for (int s = 0; s < 4; ++s) {
    int n = (tn4 * 4 + s) * 16 + (lane & 15);
    float bv = bias ? bias[n] : 0.f;
    #pragma unroll
    for (int r = 0; r < 8; ++r) {
      int m = mbase + r;
      float v = acc[s][r] + bv;
      if (fuse == 1) v = gelu_exact(v);
      if (Cadd) v += Cadd[(size_t)m * ldc + n];
      if (C)   C[(size_t)m * ldc + n] = v;
      if (Cbf) Cbf[(size_t)m * ldc + n] = (__bf16)v;
    }
  }
}

// ---------------------------------------------------------------------------
// per-(batch,head) attention: T=30, DH=64 — one wave per (b,h)
// ---------------------------------------------------------------------------
__global__ void k_attn(const float* __restrict__ q, const float* __restrict__ k,
                       const float* __restrict__ v, float* __restrict__ c,
                       __bf16* __restrict__ cbf) {
  int bh = blockIdx.x;
  int b = bh / cNH, h = bh % cNH;
  int i = threadIdx.x;
  if (i >= cT) return;
  const float* qr = q + ((size_t)(b * cT + i) * cH) + h * cDH;
  float qv[cDH];
  #pragma unroll
  for (int d = 0; d < cDH; ++d) qv[d] = qr[d];
  float sc[cT];
  float mx = -1e30f;
  for (int j = 0; j < cT; ++j) {
    const float* kr = k + ((size_t)(b * cT + j) * cH) + h * cDH;
    float s = 0.f;
    #pragma unroll
    for (int d = 0; d < cDH; ++d) s += qv[d] * kr[d];
    s *= 0.125f;                    // 1/sqrt(64)
    sc[j] = s; mx = fmaxf(mx, s);
  }
  float sum = 0.f;
  for (int j = 0; j < cT; ++j) { sc[j] = __expf(sc[j] - mx); sum += sc[j]; }
  float inv = 1.f / sum;
  float* cr = c + ((size_t)(b * cT + i) * cH) + h * cDH;
  __bf16* cbr = cbf + ((size_t)(b * cT + i) * cH) + h * cDH;
  for (int d = 0; d < cDH; ++d) {
    float acc = 0.f;
    for (int j = 0; j < cT; ++j)
      acc += sc[j] * v[((size_t)(b * cT + j) * cH) + h * cDH + d];
    acc *= inv;
    cr[d] = acc; cbr[d] = (__bf16)acc;
  }
}

// ---------------------------------------------------------------------------
// masked region pooling: pooled[b,d,:] = mean over masked regions of a[b,1+r,:]
// ---------------------------------------------------------------------------
__global__ void k_pool(const float* __restrict__ a, const int* __restrict__ mask,
                       float* __restrict__ pooled, __bf16* __restrict__ pooledb) {
  int b = blockIdx.x / cD, d = blockIdx.x % cD;
  float cnt = 0.f;
  for (int r = 0; r < cR; ++r) cnt += (mask[r * cD + d] != 0) ? 1.f : 0.f;
  float denom = 1.f / fmaxf(cnt, 1.f);
  for (int h = threadIdx.x; h < cH; h += blockDim.x) {
    float s = 0.f;
    for (int r = 0; r < cR; ++r)
      if (mask[r * cD + d] != 0)
        s += a[((size_t)b * cT + (r + 1)) * cH + h];
    s *= denom;
    size_t o = ((size_t)b * cD + d) * cH + h;
    pooled[o] = s;
    pooledb[o] = (__bf16)s;
  }
}

// ---------------------------------------------------------------------------
// classifier head: LN(K=384) -> gelu -> dot(cW2) -> gate = (cnt>0 && pred>0)
// ---------------------------------------------------------------------------
__global__ void k_cls(const float* __restrict__ h1, const float* __restrict__ cg,
                      const float* __restrict__ cbb, const float* __restrict__ cW2,
                      const float* __restrict__ cb2, const int* __restrict__ mask,
                      float* __restrict__ gate) {
  __shared__ float red[2][8];
  int b = blockIdx.x / cD, d = blockIdx.x % cD;
  const float* row = h1 + ((size_t)b * cD + d) * cK;
  float s = 0.f, ss = 0.f;
  for (int i = threadIdx.x; i < cK; i += blockDim.x) { float v = row[i]; s += v; ss += v * v; }
  for (int o = 16; o > 0; o >>= 1) { s += __shfl_xor(s, o, 32); ss += __shfl_xor(ss, o, 32); }
  int wv = threadIdx.x >> 5, ln = threadIdx.x & 31;
  if (ln == 0) { red[0][wv] = s; red[1][wv] = ss; }
  __syncthreads();
  if (threadIdx.x == 0) {
    float a = 0.f, c = 0.f;
    for (int q = 0; q < 8; ++q) { a += red[0][q]; c += red[1][q]; }
    red[0][0] = a; red[1][0] = c;
  }
  __syncthreads();
  float mean = red[0][0] / cK;
  float var  = red[1][0] / cK - mean * mean;
  float rstd = rsqrtf(var + cEPS);
  __syncthreads();
  float dotp = 0.f;
  for (int i = threadIdx.x; i < cK; i += blockDim.x) {
    float v = (row[i] - mean) * rstd * cg[(size_t)d * cK + i] + cbb[(size_t)d * cK + i];
    v = gelu_exact(v);
    dotp += v * cW2[(size_t)d * cK + i];
  }
  for (int o = 16; o > 0; o >>= 1) dotp += __shfl_xor(dotp, o, 32);
  if (ln == 0) red[0][wv] = dotp;
  __syncthreads();
  if (threadIdx.x == 0) {
    float pred = 0.f;
    for (int q = 0; q < 8; ++q) pred += red[0][q];
    pred += cb2[d];
    int cnt = 0;
    for (int r = 0; r < cR; ++r) cnt += (mask[r * cD + d] != 0);
    gate[(size_t)b * cD + d] = (cnt > 0 && pred > 0.f) ? 1.f : 0.f;
  }
}

// ---------------------------------------------------------------------------
// LoRA projection: t[row,lr] = sum_k X[srow,k] * Aw[e,lr,k]   (8 waves == LR)
// broadcastE=1: srow collapses the E axis (X indexed (b,t))
// ---------------------------------------------------------------------------
__global__ void k_lora(const __bf16* __restrict__ X, const __bf16* __restrict__ Aw,
                       float* __restrict__ t, int Ksz, int broadcastE) {
  int row = blockIdx.x;                 // over B*E*T
  int e = (row / cT) % cE;
  int srow = broadcastE ? (row / (cE * cT)) * cT + (row % cT) : row;
  int wv = threadIdx.x >> 5, lane = threadIdx.x & 31;
  const __bf16* x  = X + (size_t)srow * Ksz;
  const __bf16* aw = Aw + ((size_t)e * cLR + wv) * Ksz;
  float s = 0.f;
  for (int kk = lane; kk < Ksz; kk += 32) s += (float)x[kk] * (float)aw[kk];
  for (int o = 16; o > 0; o >>= 1) s += __shfl_xor(s, o, 32);
  if (lane == 0) t[(size_t)row * cLR + wv] = s;
}

// ---------------------------------------------------------------------------
// he[b,e,t,f] = hid[b,t,f] + SC * sum_lr t1[b,e,t,lr] * Bup[e,f,lr]  (bf16 out)
// ---------------------------------------------------------------------------
__global__ void k_he(const float* __restrict__ hid, const float* __restrict__ t1,
                     const __bf16* __restrict__ Bup, __bf16* __restrict__ he) {
  int row = blockIdx.x;                 // over B*E*T
  int e = (row / cT) % cE;
  int srow = (row / (cE * cT)) * cT + (row % cT);
  float tv[cLR];
  #pragma unroll
  for (int l = 0; l < cLR; ++l) tv[l] = t1[(size_t)row * cLR + l];
  const __bf16* bw = Bup + (size_t)e * cFF * cLR;
  for (int f = threadIdx.x; f < cFF; f += blockDim.x) {
    float s = 0.f;
    #pragma unroll
    for (int l = 0; l < cLR; ++l) s += tv[l] * (float)bw[(size_t)f * cLR + l];
    he[(size_t)row * cFF + f] = (__bf16)(hid[(size_t)srow * cFF + f] + cSC * s);
  }
}

// ---------------------------------------------------------------------------
// oe += SC * t2 @ Bdn[e]^T ; oe = ln0(oe)   (in-place, row = (b,e,t))
// ---------------------------------------------------------------------------
__global__ void k_oefin(float* __restrict__ oe, const float* __restrict__ t2,
                        const __bf16* __restrict__ Bdn) {
  __shared__ float buf[cH];
  __shared__ float red[2][8];
  int row = blockIdx.x;
  int e = (row / cT) % cE;
  float tv[cLR];
  #pragma unroll
  for (int l = 0; l < cLR; ++l) tv[l] = t2[(size_t)row * cLR + l];
  float s = 0.f, ss = 0.f;
  for (int h = threadIdx.x; h < cH; h += blockDim.x) {
    float v = oe[(size_t)row * cH + h];
    float dd = 0.f;
    #pragma unroll
    for (int l = 0; l < cLR; ++l) dd += tv[l] * (float)Bdn[((size_t)e * cH + h) * cLR + l];
    v += cSC * dd;
    buf[h] = v; s += v; ss += v * v;
  }
  for (int o = 16; o > 0; o >>= 1) { s += __shfl_xor(s, o, 32); ss += __shfl_xor(ss, o, 32); }
  int wv = threadIdx.x >> 5, ln = threadIdx.x & 31;
  if (ln == 0) { red[0][wv] = s; red[1][wv] = ss; }
  __syncthreads();
  if (threadIdx.x == 0) {
    float a = 0.f, c = 0.f;
    for (int q = 0; q < 8; ++q) { a += red[0][q]; c += red[1][q]; }
    red[0][0] = a; red[1][0] = c;
  }
  __syncthreads();
  float mean = red[0][0] / cH;
  float var  = red[1][0] / cH - mean * mean;
  float rstd = rsqrtf(var + cEPS);
  for (int h = threadIdx.x; h < cH; h += blockDim.x)
    oe[(size_t)row * cH + h] = (buf[h] - mean) * rstd;
}

// ---------------------------------------------------------------------------
// gated expert combine + ln2:
//   t==0: x = ln2(oe[b,E-1,0,:])
//   t> 0: x = ln2( (sum_d act*oe[b,d,t,:] + oe[b,E-1,t,:]) * w )
// ---------------------------------------------------------------------------
__global__ void k_combine(const float* __restrict__ oe, const float* __restrict__ gate,
                          const int* __restrict__ mask, const float* __restrict__ g2,
                          const float* __restrict__ b2, float* __restrict__ x) {
  __shared__ float buf[cH];
  __shared__ float red[2][8];
  int row = blockIdx.x;                 // over B*T
  int b = row / cT, t = row % cT;
  const float* gen = oe + (((size_t)b * cE + (cE - 1)) * cT + t) * cH;
  float w = 1.f;
  float actd[cD];
  #pragma unroll
  for (int d = 0; d < cD; ++d) actd[d] = 0.f;
  if (t > 0) {
    int r = t - 1;
    float asum = 0.f;
    for (int d = 0; d < cD; ++d) {
      float ad = (mask[r * cD + d] != 0 && gate[(size_t)b * cD + d] > 0.5f) ? 1.f : 0.f;
      actd[d] = ad; asum += ad;
    }
    w = 1.f / (asum + 1.f);
  }
  float s = 0.f, ss = 0.f;
  for (int h = threadIdx.x; h < cH; h += blockDim.x) {
    float v = gen[h];
    if (t > 0) {
      for (int d = 0; d < cD; ++d)
        if (actd[d] != 0.f)
          v += oe[(((size_t)b * cE + d) * cT + t) * cH + h];
      v *= w;
    }
    buf[h] = v; s += v; ss += v * v;
  }
  for (int o = 16; o > 0; o >>= 1) { s += __shfl_xor(s, o, 32); ss += __shfl_xor(ss, o, 32); }
  int wv = threadIdx.x >> 5, ln = threadIdx.x & 31;
  if (ln == 0) { red[0][wv] = s; red[1][wv] = ss; }
  __syncthreads();
  if (threadIdx.x == 0) {
    float a = 0.f, c = 0.f;
    for (int q = 0; q < 8; ++q) { a += red[0][q]; c += red[1][q]; }
    red[0][0] = a; red[1][0] = c;
  }
  __syncthreads();
  float mean = red[0][0] / cH;
  float var  = red[1][0] / cH - mean * mean;
  float rstd = rsqrtf(var + cEPS);
  for (int h = threadIdx.x; h < cH; h += blockDim.x)
    x[(size_t)row * cH + h] = (buf[h] - mean) * rstd * g2[h] + b2[h];
}

// ---------------------------------------------------------------------------
// host-side orchestration
// ---------------------------------------------------------------------------
extern "C" void kernel_launch(void* const* d_in, const int* in_sizes, int n_in,
                              void* d_out, int out_size, void* d_ws, size_t ws_size,
                              hipStream_t stream) {
  (void)in_sizes; (void)n_in; (void)out_size; (void)ws_size;

  const float* region = (const float*)d_in[0];
  const float* cls    = (const float*)d_in[1];
  const float* Wq  = (const float*)d_in[2];
  const float* bq  = (const float*)d_in[3];
  const float* Wk  = (const float*)d_in[4];
  const float* bk  = (const float*)d_in[5];
  const float* Wv  = (const float*)d_in[6];
  const float* bv  = (const float*)d_in[7];
  const float* Wo  = (const float*)d_in[8];
  const float* bo  = (const float*)d_in[9];
  const float* ln1g = (const float*)d_in[10];
  const float* ln1b = (const float*)d_in[11];
  const float* ln2g = (const float*)d_in[12];
  const float* ln2b = (const float*)d_in[13];
  const float* Wi  = (const float*)d_in[14];
  const float* bi  = (const float*)d_in[15];
  const float* Wof = (const float*)d_in[16];
  const float* bof = (const float*)d_in[17];
  const float* Aup = (const float*)d_in[18];
  const float* Bup = (const float*)d_in[19];
  const float* Adn = (const float*)d_in[20];
  const float* Bdn = (const float*)d_in[21];
  const float* cW1 = (const float*)d_in[22];
  const float* cb1 = (const float*)d_in[23];
  const float* cgp = (const float*)d_in[24];
  const float* cbbp = (const float*)d_in[25];
  const float* cW2 = (const float*)d_in[26];
  const float* cb2 = (const float*)d_in[27];
  const float* fing = (const float*)d_in[28];
  const float* finb = (const float*)d_in[29];
  const int*   mask = (const int*)d_in[30];
  float* out = (float*)d_out;

  // ---- workspace carving (deterministic bump allocator) ----
  char* wp = (char*)d_ws;
  auto carve = [&](size_t bytes) -> char* {
    char* r = wp;
    wp += (bytes + 255) & ~(size_t)255;
    return r;
  };

  float*  xbuf  = (float*) carve((size_t)cM * cH * 4);
  __bf16* xnb   = (__bf16*)carve((size_t)cM * cH * 2);
  float*  qf    = (float*) carve((size_t)cM * cH * 4);
  float*  kf    = (float*) carve((size_t)cM * cH * 4);
  float*  vf    = (float*) carve((size_t)cM * cH * 4);
  float*  cf    = (float*) carve((size_t)cM * cH * 4);
  __bf16* cb_   = (__bf16*)carve((size_t)cM * cH * 2);
  float*  af    = (float*) carve((size_t)cM * cH * 4);
  __bf16* ab    = (__bf16*)carve((size_t)cM * cH * 2);
  float*  hidf  = (float*) carve((size_t)cM * cFF * 4);
  __bf16* hidb  = (__bf16*)carve((size_t)cM * cFF * 2);
  float*  pooledf = (float*) carve((size_t)cB * cD * cH * 4);
  __bf16* pooledb = (__bf16*)carve((size_t)cB * cD * cH * 2);
  float*  h1f   = (float*) carve((size_t)cB * cD * cK * 4);
  float*  gate  = (float*) carve((size_t)cB * cD * 4);
  float*  t1    = (float*) carve((size_t)cBET * cLR * 4);
  float*  t2    = (float*) carve((size_t)cBET * cLR * 4);
  __bf16* heb   = (__bf16*)carve((size_t)cBET * cFF * 2);
  float*  oef   = (float*) carve((size_t)cBET * cH * 4);

  // bf16 weight mirrors
  __bf16* WqB  = (__bf16*)carve((size_t)cL * cH * cH * 2);
  __bf16* WkB  = (__bf16*)carve((size_t)cL * cH * cH * 2);
  __bf16* WvB  = (__bf16*)carve((size_t)cL * cH * cH * 2);
  __bf16* WoB  = (__bf16*)carve((size_t)cL * cH * cH * 2);
  __bf16* WiB  = (__bf16*)carve((size_t)cL * cFF * cH * 2);
  __bf16* WofB = (__bf16*)carve((size_t)cL * cH * cFF * 2);
  __bf16* AupB = (__bf16*)carve((size_t)(cL/2) * cE * cLR * cH * 2);
  __bf16* BupB = (__bf16*)carve((size_t)(cL/2) * cE * cFF * cLR * 2);
  __bf16* AdnB = (__bf16*)carve((size_t)(cL/2) * cE * cLR * cFF * 2);
  __bf16* BdnB = (__bf16*)carve((size_t)(cL/2) * cE * cH * cLR * 2);
  __bf16* cW1B = (__bf16*)carve((size_t)(cL/2) * cD * cK * cH * 2);

  auto cvt = [&](const float* src, __bf16* dst, long n) {
    long blocks = (n + 255) / 256;
    if (blocks > 16384) blocks = 16384;
    k_cvt<<<dim3((unsigned)blocks), dim3(256), 0, stream>>>(src, dst, n);
  };
  cvt(Wq,  WqB,  (long)cL * cH * cH);
  cvt(Wk,  WkB,  (long)cL * cH * cH);
  cvt(Wv,  WvB,  (long)cL * cH * cH);
  cvt(Wo,  WoB,  (long)cL * cH * cH);
  cvt(Wi,  WiB,  (long)cL * cFF * cH);
  cvt(Wof, WofB, (long)cL * cH * cFF);
  cvt(Aup, AupB, (long)(cL/2) * cE * cLR * cH);
  cvt(Bup, BupB, (long)(cL/2) * cE * cFF * cLR);
  cvt(Adn, AdnB, (long)(cL/2) * cE * cLR * cFF);
  cvt(Bdn, BdnB, (long)(cL/2) * cE * cH * cLR);
  cvt(cW1, cW1B, (long)(cL/2) * cD * cK * cH);

  auto gemm = [&](const __bf16* A, int lda, const __bf16* W, int ldb,
                  const float* bias, const float* Cadd, float* C, __bf16* Cbf,
                  int ldc, int M, int N, int Ksz, int fuse) {
    int Mt = M / 16, Nt4 = N / 64;
    int waves = Mt * Nt4;
    int blocks = (waves + 7) / 8;
    k_gemm<<<dim3(blocks), dim3(256), 0, stream>>>(A, lda, W, ldb, bias, Cadd,
                                                   C, Cbf, ldc, Mt, Nt4, Ksz, fuse);
  };

  // x = concat(cls, region)
  k_concat<<<dim3(cM), dim3(256), 0, stream>>>(region, cls, xbuf);

  for (int i = 0; i < cL; ++i) {
    // xn = ln1(x)  (bf16 feed for WMMA)
    k_ln<<<dim3(cM), dim3(256), 0, stream>>>(xbuf, ln1g + (size_t)i * cH,
                                             ln1b + (size_t)i * cH, nullptr, xnb, cH);
    // q/k/v projections
    gemm(xnb, cH, WqB + (size_t)i * cH * cH, cH, bq + (size_t)i * cH,
         nullptr, qf, nullptr, cH, cM, cH, cH, 0);
    gemm(xnb, cH, WkB + (size_t)i * cH * cH, cH, bk + (size_t)i * cH,
         nullptr, kf, nullptr, cH, cM, cH, cH, 0);
    gemm(xnb, cH, WvB + (size_t)i * cH * cH, cH, bv + (size_t)i * cH,
         nullptr, vf, nullptr, cH, cM, cH, cH, 0);
    // attention
    k_attn<<<dim3(cB * cNH), dim3(32), 0, stream>>>(qf, kf, vf, cf, cb_);

    if ((i & 1) == 0) {
      int ei = i / 2;
      // a = ctx @ Wo^T + bo   (fp32 + bf16 mirror)
      gemm(cb_, cH, WoB + (size_t)i * cH * cH, cH, bo + (size_t)i * cH,
           nullptr, af, ab, cH, cM, cH, cH, 0);
      // pooled tissue features + classifier
      k_pool<<<dim3(cB * cD), dim3(256), 0, stream>>>(af, mask, pooledf, pooledb);
      for (int d = 0; d < cD; ++d) {
        gemm(pooledb + (size_t)d * cH, cD * cH,
             cW1B + ((size_t)ei * cD + d) * cK * cH, cH,
             cb1 + ((size_t)ei * cD + d) * cK,
             nullptr, h1f + (size_t)d * cK, nullptr, cD * cK, cB, cK, cH, 0);
      }
      k_cls<<<dim3(cB * cD), dim3(256), 0, stream>>>(
          h1f, cgp + (size_t)ei * cD * cK, cbbp + (size_t)ei * cD * cK,
          cW2 + (size_t)ei * cD * cK, cb2 + (size_t)ei * cD, mask, gate);
      // hid = gelu(a @ Wi^T + bi)
      gemm(ab, cH, WiB + (size_t)i * cFF * cH, cH, bi + (size_t)i * cFF,
           nullptr, hidf, nullptr, cFF, cM, cFF, cH, 1);
      // LoRA up: t1 = a @ Aup^T ; he = hid + SC * t1 @ Bup^T
      k_lora<<<dim3(cBET), dim3(256), 0, stream>>>(
          ab, AupB + (size_t)ei * cE * cLR * cH, t1, cH, 1);
      k_he<<<dim3(cBET), dim3(256), 0, stream>>>(
          hidf, t1, BupB + (size_t)ei * cE * cFF * cLR, heb);
      // oe = he @ Wof^T + bof  (single flat GEMM over B*E*T rows)
      gemm(heb, cFF, WofB + (size_t)i * cH * cFF, cFF, bof + (size_t)i * cH,
           nullptr, oef, nullptr, cH, cBET, cH, cFF, 0);
      // LoRA down + ln0 (in place on oe)
      k_lora<<<dim3(cBET), dim3(256), 0, stream>>>(
          heb, AdnB + (size_t)ei * cE * cLR * cFF, t2, cFF, 0);
      k_oefin<<<dim3(cBET), dim3(256), 0, stream>>>(
          oef, t2, BdnB + (size_t)ei * cE * cH * cLR);
      // gated combine + ln2 -> x
      k_combine<<<dim3(cM), dim3(256), 0, stream>>>(
          oef, gate, mask, ln2g + (size_t)i * cH, ln2b + (size_t)i * cH, xbuf);
    } else {
      // x = x + (ctx @ Wo^T + bo)    (residual fused into GEMM epilogue)
      gemm(cb_, cH, WoB + (size_t)i * cH * cH, cH, bo + (size_t)i * cH,
           xbuf, xbuf, nullptr, cH, cM, cH, cH, 0);
      // xn2 = ln2(x)
      k_ln<<<dim3(cM), dim3(256), 0, stream>>>(xbuf, ln2g + (size_t)i * cH,
                                               ln2b + (size_t)i * cH, nullptr, xnb, cH);
      // x = x + gelu(xn2 @ Wi^T + bi) @ Wof^T + bof
      gemm(xnb, cH, WiB + (size_t)i * cFF * cH, cH, bi + (size_t)i * cFF,
           nullptr, nullptr, hidb, cFF, cM, cFF, cH, 1);
      gemm(hidb, cFF, WofB + (size_t)i * cH * cFF, cFF, bof + (size_t)i * cH,
           xbuf, xbuf, nullptr, cH, cM, cH, cFF, 0);
    }
  }

  // final LayerNorm -> output
  k_ln<<<dim3(cM), dim3(256), 0, stream>>>(xbuf, fing, finb, out, nullptr, cH);
}
